// EquivariantAttention_87557203296733
// MI455X (gfx1250) — compile-verified, matrix operand order checked by reading, not run on hardware
//
#include <hip/hip_runtime.h>
#include <hip/hip_fp16.h>
#include <math.h>

#define B_   2
#define N_   4096
#define DIM_ 256
#define H_   4
#define DH_  64
#define NN_  32
#define QTOT (B_*N_)   /* 8192 query rows */

typedef __attribute__((ext_vector_type(16))) _Float16 v16h;
typedef __attribute__((ext_vector_type(8)))  _Float16 v8h;
typedef __attribute__((ext_vector_type(8)))  float    v8f;

// ---------------------------------------------------------------- utilities
__device__ __forceinline__ float wsum(float v) {
#pragma unroll
  for (int o = 16; o > 0; o >>= 1) v += __shfl_xor(v, o, 32);
  return v;
}

// Load a 16x32 f16 fragment (A-layout) from row-major storage with leading
// dim `ld`. Per ISA: lane L holds row (L&15); halves 0..7 -> K = hi*8 + e,
// halves 8..15 -> K = hi*8 + 16 + e, hi = L>>4.  Two 16B contiguous loads.
// B fragments use the same loader on pre-TRANSPOSED weights (N rows x K).
__device__ __forceinline__ v16h load_frag16(const _Float16* base, int ld, int lane) {
  int r = lane & 15, hi = lane >> 4;
  const _Float16* p = base + (size_t)r * ld + hi * 8;
  v8h lo = *(const v8h*)p;
  v8h hv = *(const v8h*)(p + 16);
  v16h out;
#pragma unroll
  for (int e = 0; e < 8; ++e) { out[e] = lo[e]; out[e + 8] = hv[e]; }
  return out;
}

__device__ __forceinline__ float siluf(float y) { return y / (1.f + expf(-y)); }
__device__ __forceinline__ float geluf(float x) { return 0.5f * x * (1.f + erff(x * 0.70710678118f)); }

// ------------------------------------------------- weight convert+transpose
// W is K x N row-major fp32; WT is N x K row-major f16.
__global__ void cvt_wT_kernel(const float* __restrict__ W, _Float16* __restrict__ WT,
                              int K, int N) {
  int o = blockIdx.x * 256 + threadIdx.x;
  if (o >= K * N) return;
  int n = o / K, k = o % K;
  WT[o] = (_Float16)W[k * N + n];
}

// ------------------------------------------------------------- layernorm(x)
// one wave per 256-wide row, 8 rows per 256-thread block
__global__ void ln_kernel(const float* __restrict__ feats, const float* __restrict__ gamma,
                          _Float16* __restrict__ xh) {
  int row  = blockIdx.x * 8 + (threadIdx.x >> 5);
  int lane = threadIdx.x & 31;
  if (row >= QTOT) return;
  const float* xr = feats + (size_t)row * DIM_;
  float v[8], s = 0.f;
#pragma unroll
  for (int e = 0; e < 8; ++e) { v[e] = xr[lane + 32 * e]; s += v[e]; }
  s = wsum(s);
  float m = s * (1.f / 256.f), vs = 0.f;
#pragma unroll
  for (int e = 0; e < 8; ++e) { float d = v[e] - m; vs += d * d; }
  vs = wsum(vs);
  float inv = rsqrtf(vs * (1.f / 256.f) + 1e-5f);
#pragma unroll
  for (int e = 0; e < 8; ++e) {
    int d = lane + 32 * e;
    xh[(size_t)row * DIM_ + d] = (_Float16)((v[e] - m) * inv * gamma[d]);
  }
}

// ------------------------------------------------------------ generic GEMM
// C(MxN,f32) = A(MxK,f16 row-major) * B(KxN) with B given TRANSPOSED (NxK f16)
// Register-blocked 1x4: one A fragment reused across 4 B tiles per k-step.
// One 16x64 output strip per wave, 4 waves per block. N must be mult of 64.
__global__ void gemm_kernel(const _Float16* __restrict__ A, const _Float16* __restrict__ BT,
                            const float* __restrict__ bias, float* __restrict__ C,
                            int M, int N, int K) {
  int lane = threadIdx.x & 31;
  int wid  = blockIdx.x * 4 + (threadIdx.x >> 5);
  int ng   = N >> 6;                       // groups of 4 n-tiles
  if (wid >= (M >> 4) * ng) return;
  int mt = wid / ng, n0 = (wid % ng) * 4;  // 4 consecutive 16-wide n tiles
  const _Float16* Arow = A + (size_t)(mt * 16) * K;
  v8f acc[4] = {};
  for (int k0 = 0; k0 < K; k0 += 32) {
    // prefetch next k-block of the streamed A operand (global_prefetch_b8)
    if (k0 + 32 < K)
      __builtin_prefetch(Arow + (size_t)(lane & 15) * K + k0 + 32, 0, 1);
    v16h a = load_frag16(Arow + k0, K, lane);
#pragma unroll
    for (int u = 0; u < 4; ++u) {
      v16h b = load_frag16(BT + (size_t)((n0 + u) * 16) * K + k0, K, lane);
      acc[u] = __builtin_amdgcn_wmma_f32_16x16x32_f16(false, a, false, b, (short)0, acc[u], false, false);
    }
  }
  int n = lane & 15, hi = lane >> 4;
#pragma unroll
  for (int u = 0; u < 4; ++u) {
    int col = (n0 + u) * 16 + n;
    float bv = bias ? bias[col] : 0.f;
#pragma unroll
    for (int r = 0; r < 8; ++r)
      C[(size_t)(mt * 16 + hi * 8 + r) * N + col] = acc[u][r] + bv;
  }
}

// -------------------------------------------------------------------- KNN
// one block (256 threads) per query; LDS distance table + 32 argmin rounds
__global__ void knn_kernel(const float* __restrict__ coors, int* __restrict__ nidx,
                           float* __restrict__ ndist, float* __restrict__ nrel) {
  __shared__ float dist[N_];
  __shared__ float rmin[256];
  __shared__ int   rarg[256];
  int q = blockIdx.x, b = q / N_, i = q % N_;
  const float* cb = coors + (size_t)b * N_ * 3;
  float cx = cb[i * 3 + 0], cy = cb[i * 3 + 1], cz = cb[i * 3 + 2];
  int t = threadIdx.x;
  for (int j = t; j < N_; j += 256) {
    float dx = cx - cb[j * 3 + 0], dy = cy - cb[j * 3 + 1], dz = cz - cb[j * 3 + 2];
    dist[j] = dx * dx + dy * dy + dz * dz;
  }
  __syncthreads();
  for (int r = 0; r < NN_; ++r) {
    float best = 3.4e38f; int barg = N_;
    for (int j = t; j < N_; j += 256) {
      float d = dist[j];
      if (d < best || (d == best && j < barg)) { best = d; barg = j; }
    }
    rmin[t] = best; rarg[t] = barg;
    __syncthreads();
    for (int s = 128; s > 0; s >>= 1) {
      if (t < s) {
        if (rmin[t + s] < rmin[t] || (rmin[t + s] == rmin[t] && rarg[t + s] < rarg[t])) {
          rmin[t] = rmin[t + s]; rarg[t] = rarg[t + s];
        }
      }
      __syncthreads();
    }
    if (t == 0) {
      int jm = rarg[0];
      nidx[q * NN_ + r]  = jm;
      ndist[q * NN_ + r] = sqrtf(dist[jm]);
      float dx = cx - cb[jm * 3 + 0], dy = cy - cb[jm * 3 + 1], dz = cz - cb[jm * 3 + 2];
      nrel[(size_t)(q * NN_ + r) * 3 + 0] = dx;
      nrel[(size_t)(q * NN_ + r) * 3 + 1] = dy;
      nrel[(size_t)(q * NN_ + r) * 3 + 2] = dz;
      dist[jm] = 3.4e38f;
    }
    __syncthreads();
  }
}

// ---------------------------------------------------- fused DPB MLP (WMMA)
__device__ void dpb_layer(_Float16* ph, float* buf, const _Float16* WT,
                          const float* bb, const float* gg, const float* be,
                          int lane, int wave) {
  // buf(32x128) = ph(32x128) @ W(128x128)  -- WT transposed.
  // wave -> (mt = wave>>1, 4 n-tiles); A k-fragments hoisted and reused.
  {
    int mt = wave >> 1, nbase = (wave & 1) * 4;
    v16h a0 = load_frag16(ph + (mt * 16) * 128 + 0,  128, lane);
    v16h a1 = load_frag16(ph + (mt * 16) * 128 + 32, 128, lane);
    v16h a2 = load_frag16(ph + (mt * 16) * 128 + 64, 128, lane);
    v16h a3 = load_frag16(ph + (mt * 16) * 128 + 96, 128, lane);
#pragma unroll
    for (int u = 0; u < 4; ++u) {
      int nt = nbase + u;
      const _Float16* Wn = WT + (size_t)(nt * 16) * 128;
      v8f acc = {};
      acc = __builtin_amdgcn_wmma_f32_16x16x32_f16(false, a0, false, load_frag16(Wn + 0,  128, lane), (short)0, acc, false, false);
      acc = __builtin_amdgcn_wmma_f32_16x16x32_f16(false, a1, false, load_frag16(Wn + 32, 128, lane), (short)0, acc, false, false);
      acc = __builtin_amdgcn_wmma_f32_16x16x32_f16(false, a2, false, load_frag16(Wn + 64, 128, lane), (short)0, acc, false, false);
      acc = __builtin_amdgcn_wmma_f32_16x16x32_f16(false, a3, false, load_frag16(Wn + 96, 128, lane), (short)0, acc, false, false);
      int n = lane & 15, hi = lane >> 4;
#pragma unroll
      for (int r = 0; r < 8; ++r) buf[(mt * 16 + hi * 8 + r) * 128 + nt * 16 + n] = acc[r];
    }
  }
  __syncthreads();
  // LN + SiLU per row (wave handles 8 rows; lane covers 4 dims each)
  for (int jj = 0; jj < 8; ++jj) {
    int j = wave * 8 + jj;
    float v[4], s = 0.f;
#pragma unroll
    for (int e = 0; e < 4; ++e) { int d = lane + 32 * e; v[e] = buf[j * 128 + d] + bb[d]; s += v[e]; }
    s = wsum(s); float m = s * (1.f / 128.f), vs = 0.f;
#pragma unroll
    for (int e = 0; e < 4; ++e) { float dd = v[e] - m; vs += dd * dd; }
    vs = wsum(vs); float inv = rsqrtf(vs * (1.f / 128.f) + 1e-5f);
#pragma unroll
    for (int e = 0; e < 4; ++e) {
      int d = lane + 32 * e;
      ph[j * 128 + d] = (_Float16)siluf((v[e] - m) * inv * gg[d] + be[d]);
    }
  }
  __syncthreads();
}

__global__ void dpb_kernel(const float* __restrict__ ndist,
                           const float* w0, const float* b0, const float* g0, const float* be0,
                           const _Float16* w1T, const float* b1, const float* g1, const float* be1,
                           const _Float16* w2T, const float* b2, const float* g2, const float* be2,
                           const float* qkw, const float* qkb,
                           _Float16* __restrict__ posf, float* __restrict__ qkpos) {
  __shared__ float    buf[32 * 128];
  __shared__ _Float16 ph[32 * 128];
  int q = blockIdx.x, t = threadIdx.x, lane = t & 31, wave = t >> 5;
  // layer 0: scalar (-dist) -> 128 + LN + SiLU
  for (int jj = 0; jj < 8; ++jj) {
    int j = wave * 8 + jj;
    float x0 = -ndist[q * NN_ + j];
    float v[4], s = 0.f;
#pragma unroll
    for (int e = 0; e < 4; ++e) { int d = lane + 32 * e; v[e] = x0 * w0[d] + b0[d]; s += v[e]; }
    s = wsum(s); float m = s * (1.f / 128.f), vs = 0.f;
#pragma unroll
    for (int e = 0; e < 4; ++e) { float dd = v[e] - m; vs += dd * dd; }
    vs = wsum(vs); float inv = rsqrtf(vs * (1.f / 128.f) + 1e-5f);
#pragma unroll
    for (int e = 0; e < 4; ++e) {
      int d = lane + 32 * e;
      ph[j * 128 + d] = (_Float16)siluf((v[e] - m) * inv * g0[d] + be0[d]);
    }
  }
  __syncthreads();
  dpb_layer(ph, buf, w1T, b1, g1, be1, lane, wave);
  dpb_layer(ph, buf, w2T, b2, g2, be2, lane, wave);
  // export final pos activations (f16) for the attention kernel
  for (int e = t; e < 32 * 128; e += 128) posf[(size_t)q * 4096 + e] = ph[e];
  // qk position bias: 128 (h,j) dots of length 128
  int h = t & 3, j = t >> 2;
  float s = 0.f;
  for (int d = 0; d < 128; ++d) s += (float)ph[j * 128 + d] * qkw[d * H_ + h];
  qkpos[(size_t)q * (H_ * NN_) + h * NN_ + j] = s + qkb[h];
}

// ------------------------------------------------- fused attention (WMMA)
__global__ void attn_kernel(const float* __restrict__ qkv, const int* __restrict__ nidx,
                            const float* __restrict__ nrel, const float* __restrict__ qkpos,
                            const _Float16* __restrict__ posf, const _Float16* __restrict__ vposT,
                            const float* __restrict__ vposb,
                            const float* __restrict__ cmw1, const float* __restrict__ cmw2,
                            const float* __restrict__ gatew, const float* __restrict__ gateb,
                            const float* __restrict__ cns, const float* __restrict__ combine,
                            _Float16* __restrict__ outpre, float* __restrict__ coors_out) {
  __shared__ _Float16 ks[NN_][DIM_];
  __shared__ _Float16 vs_[NN_][DIM_];
  __shared__ _Float16 ph[NN_ * 128];
  __shared__ float    qs[DIM_];
  __shared__ float    qk_s[H_][NN_];
  __shared__ float    attn_s[H_][NN_];
  __shared__ float    cattn[H_][NN_];
  __shared__ float    sign_s[NN_][H_];
  __shared__ float    reln[NN_][3];
  __shared__ int      idx_s[NN_];
  int q = blockIdx.x, b = q >> 12;
  int t = threadIdx.x, lane = t & 31, wave = t >> 5;

  if (t < NN_) idx_s[t] = nidx[q * NN_ + t];
  __syncthreads();
  // gather q / neighbor k,v
  const float* qrow = qkv + (size_t)q * 768;
  for (int e = t; e < DIM_; e += 128) qs[e] = qrow[e];
  for (int e = t; e < NN_ * DIM_; e += 128) {
    int j = e >> 8, d = e & 255;
    const float* krow = qkv + (size_t)(b * N_ + idx_s[j]) * 768;
    ks[j][d]  = (_Float16)krow[256 + d];
    vs_[j][d] = (_Float16)krow[512 + d];
  }
  for (int e = t; e < NN_ * 128; e += 128) ph[e] = posf[(size_t)q * 4096 + e];
  __syncthreads();
  // l2norm q (wave = head, lane covers 2 of 64 dims)
  {
    int h = wave;
    float v0 = qs[h * 64 + lane], v1 = qs[h * 64 + 32 + lane];
    float s = wsum(v0 * v0 + v1 * v1);
    float inv = 1.f / fmaxf(sqrtf(s), 1e-12f);
    qs[h * 64 + lane] = v0 * inv;
    qs[h * 64 + 32 + lane] = v1 * inv;
  }
  __syncthreads();
  // l2norm k per (j,h)
  {
    int j = t >> 2, h = t & 3; float s = 0.f;
    for (int d = 0; d < 64; ++d) { float kv = (float)ks[j][h * 64 + d]; s += kv * kv; }
    float inv = 1.f / fmaxf(sqrtf(s), 1e-12f);
    for (int d = 0; d < 64; ++d) ks[j][h * 64 + d] = (_Float16)((float)ks[j][h * 64 + d] * inv);
  }
  __syncthreads();
  // qk logits (+pos bias)
  {
    int j = t >> 2, h = t & 3; float s = 0.f;
    for (int d = 0; d < 64; ++d) s += qs[h * 64 + d] * (float)ks[j][h * 64 + d];
    qk_s[h][j] = s * 8.0f + qkpos[(size_t)q * (H_ * NN_) + h * NN_ + j];
  }
  __syncthreads();
  // v_pos = pos(32x128) @ Wv(128x256), accumulated into gathered V (+bias).
  // wave -> (mt = wave>>1, 8 n-tiles); A k-fragments hoisted and reused.
  {
    int mt = wave >> 1, nbase = (wave & 1) * 8;
    v16h a0 = load_frag16(ph + (mt * 16) * 128 + 0,  128, lane);
    v16h a1 = load_frag16(ph + (mt * 16) * 128 + 32, 128, lane);
    v16h a2 = load_frag16(ph + (mt * 16) * 128 + 64, 128, lane);
    v16h a3 = load_frag16(ph + (mt * 16) * 128 + 96, 128, lane);
#pragma unroll
    for (int u = 0; u < 8; ++u) {
      int nt = nbase + u;
      const _Float16* Wn = vposT + (size_t)(nt * 16) * 128;
      v8f acc = {};
      acc = __builtin_amdgcn_wmma_f32_16x16x32_f16(false, a0, false, load_frag16(Wn + 0,  128, lane), (short)0, acc, false, false);
      acc = __builtin_amdgcn_wmma_f32_16x16x32_f16(false, a1, false, load_frag16(Wn + 32, 128, lane), (short)0, acc, false, false);
      acc = __builtin_amdgcn_wmma_f32_16x16x32_f16(false, a2, false, load_frag16(Wn + 64, 128, lane), (short)0, acc, false, false);
      acc = __builtin_amdgcn_wmma_f32_16x16x32_f16(false, a3, false, load_frag16(Wn + 96, 128, lane), (short)0, acc, false, false);
      int n = lane & 15, hi = lane >> 4, col = nt * 16 + n;
#pragma unroll
      for (int r = 0; r < 8; ++r) {
        int m = mt * 16 + hi * 8 + r;
        vs_[m][col] = (_Float16)((float)vs_[m][col] + acc[r] + vposb[col]);
      }
    }
  }
  __syncthreads();
  // coordinate branch: per-neighbor MLP 4->16->4 (gelu), gate, CoorsNorm
  if (t < NN_) {
    int j = t;
    float qv[4];
#pragma unroll
    for (int h = 0; h < 4; ++h) qv[h] = qk_s[h][j];
    float hid[16];
#pragma unroll
    for (int c = 0; c < 16; ++c) {
      float s = 0.f;
#pragma unroll
      for (int h = 0; h < 4; ++h) s += qv[h] * cmw1[h * 16 + c];
      hid[c] = geluf(s);
    }
#pragma unroll
    for (int hp = 0; hp < 4; ++hp) {
      float s = 0.f;
#pragma unroll
      for (int c = 0; c < 16; ++c) s += hid[c] * cmw2[c * 4 + hp];
      cattn[hp][j] = s;
      float g = 0.f;
#pragma unroll
      for (int h = 0; h < 4; ++h) g += qv[h] * gatew[h * 4 + hp];
      sign_s[j][hp] = tanhf(g + gateb[hp]);
    }
    float rx = nrel[(size_t)(q * NN_ + j) * 3 + 0];
    float ry = nrel[(size_t)(q * NN_ + j) * 3 + 1];
    float rz = nrel[(size_t)(q * NN_ + j) * 3 + 2];
    float inv = cns[0] / fmaxf(sqrtf(rx * rx + ry * ry + rz * rz), 1e-8f);
    reln[j][0] = rx * inv; reln[j][1] = ry * inv; reln[j][2] = rz * inv;
  }
  __syncthreads();
  // softmaxes: coor-attn over neighbors, attn over neighbors (per head)
  if (t < H_) {
    int h = t;
    float mx = -3.4e38f;
    for (int j = 0; j < NN_; ++j) mx = fmaxf(mx, cattn[h][j]);
    float s = 0.f;
    for (int j = 0; j < NN_; ++j) { float e = expf(cattn[h][j] - mx); cattn[h][j] = e; s += e; }
    float is = 1.f / s;
    for (int j = 0; j < NN_; ++j) cattn[h][j] *= is;
    mx = -3.4e38f;
    for (int j = 0; j < NN_; ++j) mx = fmaxf(mx, qk_s[h][j]);
    s = 0.f;
    for (int j = 0; j < NN_; ++j) { float e = expf(qk_s[h][j] - mx); attn_s[h][j] = e; s += e; }
    is = 1.f / s;
    for (int j = 0; j < NN_; ++j) attn_s[h][j] *= is;
  }
  __syncthreads();
  // coors_out[c] = sum_j sum_h cattn*sign*combine * rel_n[j][c]
  if (t < NN_) {
    int j = t;
    float g = 0.f;
#pragma unroll
    for (int h = 0; h < 4; ++h) g += cattn[h][j] * sign_s[j][h] * combine[h];
    float cx = wsum(g * reln[j][0]);
    float cy = wsum(g * reln[j][1]);
    float cz = wsum(g * reln[j][2]);
    if (lane == 0) {
      coors_out[(size_t)q * 3 + 0] = cx;
      coors_out[(size_t)q * 3 + 1] = cy;
      coors_out[(size_t)q * 3 + 2] = cz;
    }
  }
  __syncthreads();
  // out_pre[col] = sum_j attn[h][j] * v[j][col]
#pragma unroll
  for (int cc = 0; cc < 2; ++cc) {
    int col = t + cc * 128, h = col >> 6;
    float s = 0.f;
    for (int j = 0; j < NN_; ++j) s += attn_s[h][j] * (float)vs_[j][col];
    outpre[(size_t)q * DIM_ + col] = (_Float16)s;
  }
}

// ------------------------------------------------------------------- host
extern "C" void kernel_launch(void* const* d_in, const int* in_sizes, int n_in,
                              void* d_out, int out_size, void* d_ws, size_t ws_size,
                              hipStream_t stream) {
  (void)in_sizes; (void)n_in; (void)out_size; (void)ws_size;
  const float* feats    = (const float*)d_in[0];
  const float* coors    = (const float*)d_in[1];
  const float* gamma    = (const float*)d_in[2];
  const float* w_qkv    = (const float*)d_in[3];
  const float* w_out    = (const float*)d_in[4];
  const float* b_out    = (const float*)d_in[5];
  const float* dpb_w0   = (const float*)d_in[6];
  const float* dpb_b0   = (const float*)d_in[7];
  const float* dpb_g0   = (const float*)d_in[8];
  const float* dpb_be0  = (const float*)d_in[9];
  const float* dpb_w1   = (const float*)d_in[10];
  const float* dpb_b1   = (const float*)d_in[11];
  const float* dpb_g1   = (const float*)d_in[12];
  const float* dpb_be1  = (const float*)d_in[13];
  const float* dpb_w2   = (const float*)d_in[14];
  const float* dpb_b2   = (const float*)d_in[15];
  const float* dpb_g2   = (const float*)d_in[16];
  const float* dpb_be2  = (const float*)d_in[17];
  const float* qk_pos_w = (const float*)d_in[18];
  const float* qk_pos_b = (const float*)d_in[19];
  const float* v_pos_w  = (const float*)d_in[20];
  const float* v_pos_b  = (const float*)d_in[21];
  const float* cmw1     = (const float*)d_in[22];
  const float* cmw2     = (const float*)d_in[23];
  const float* gatew    = (const float*)d_in[24];
  const float* gateb    = (const float*)d_in[25];
  const float* cns      = (const float*)d_in[26];
  const float* combine  = (const float*)d_in[27];

  char* ws = (char*)d_ws;
  size_t off = 0;
  auto take = [&](size_t bytes) -> void* {
    void* p = ws + off;
    off = (off + bytes + 255) & ~(size_t)255;
    return p;
  };
  _Float16* xh     = (_Float16*)take((size_t)QTOT * DIM_ * 2);
  _Float16* wqkvT  = (_Float16*)take((size_t)768 * 256 * 2);
  _Float16* woutT  = (_Float16*)take((size_t)256 * 256 * 2);
  _Float16* w1T    = (_Float16*)take((size_t)128 * 128 * 2);
  _Float16* w2T    = (_Float16*)take((size_t)128 * 128 * 2);
  _Float16* vposT  = (_Float16*)take((size_t)256 * 128 * 2);
  float*    qkvb   = (float*)take((size_t)QTOT * 768 * 4);
  int*      nidx   = (int*)take((size_t)QTOT * NN_ * 4);
  float*    ndist  = (float*)take((size_t)QTOT * NN_ * 4);
  float*    nrel   = (float*)take((size_t)QTOT * NN_ * 3 * 4);
  float*    qkpos  = (float*)take((size_t)QTOT * H_ * NN_ * 4);
  _Float16* posf   = (_Float16*)take((size_t)QTOT * NN_ * 128 * 2);
  _Float16* outpre = (_Float16*)take((size_t)QTOT * DIM_ * 2);

  // weight conversion (fp32 -> f16, transposed for contiguous B fragments)
  cvt_wT_kernel<<<(256 * 768 + 255) / 256, 256, 0, stream>>>(w_qkv, wqkvT, 256, 768);
  cvt_wT_kernel<<<(256 * 256 + 255) / 256, 256, 0, stream>>>(w_out, woutT, 256, 256);
  cvt_wT_kernel<<<(128 * 128 + 255) / 256, 256, 0, stream>>>(dpb_w1, w1T, 128, 128);
  cvt_wT_kernel<<<(128 * 128 + 255) / 256, 256, 0, stream>>>(dpb_w2, w2T, 128, 128);
  cvt_wT_kernel<<<(128 * 256 + 255) / 256, 256, 0, stream>>>(v_pos_w, vposT, 128, 256);

  // pipeline
  ln_kernel<<<QTOT / 8, 256, 0, stream>>>(feats, gamma, xh);
  gemm_kernel<<<(QTOT / 16) * (768 / 64) / 4, 128, 0, stream>>>(xh, wqkvT, nullptr, qkvb, QTOT, 768, 256);
  knn_kernel<<<QTOT, 256, 0, stream>>>(coors, nidx, ndist, nrel);
  dpb_kernel<<<QTOT, 128, 0, stream>>>(ndist,
                                       dpb_w0, dpb_b0, dpb_g0, dpb_be0,
                                       w1T, dpb_b1, dpb_g1, dpb_be1,
                                       w2T, dpb_b2, dpb_g2, dpb_be2,
                                       qk_pos_w, qk_pos_b, posf, qkpos);
  float* coors_out = (float*)d_out + (size_t)QTOT * DIM_;
  attn_kernel<<<QTOT, 128, 0, stream>>>(qkvb, nidx, nrel, qkpos, posf, vposT, v_pos_b,
                                        cmw1, cmw2, gatew, gateb, cns, combine,
                                        outpre, coors_out);
  gemm_kernel<<<(QTOT / 16) * (256 / 64) / 4, 128, 0, stream>>>(outpre, woutT, b_out, (float*)d_out, QTOT, 256, 256);
}
